// DotProductAttention_28750511079851
// MI455X (gfx1250) — matrix operand order repaired and checked by reading, not compiled, not run on hardware
//
#include <hip/hip_runtime.h>
#include <math.h>

// Flash attention for B=16, Q=K=2048, D=128 (fp32 in/out) on gfx1250.
// GEMM1 computes S^T = K_tile * Q^T so the S^T C-fragment layout coincides
// exactly with the P A-fragment layout needed for GEMM2 (P * V).
// Query-row masking is folded into the Q fragment (zeroed columns), and the
// softmax runs in the log2 domain (native v_exp_f32 via exp2).
// Staging uses uniform-base + 32-bit lane-offset addressing (GVS mode) so the
// scalar unit carries the per-chunk pointer advance.

typedef _Float16 v16h __attribute__((ext_vector_type(16)));
typedef _Float16 h8   __attribute__((ext_vector_type(8)));
typedef float    v8f  __attribute__((ext_vector_type(8)));

#define BQ   64          // queries per workgroup (4 waves x 16)
#define CK   32          // keys per chunk
#define DH   128         // head dim
#define KST  136         // K LDS row stride in halves (128 + 8 pad)
#define VST  40          // V^T LDS row stride in halves (32 + 8 pad)

union F16F { v16h v; h8 h[2]; };

__global__ __launch_bounds__(128)
void attn_fwd_wmma(const float* __restrict__ Qm,
                   const float* __restrict__ Km,
                   const float* __restrict__ Vm,
                   const int*   __restrict__ valid_lens,
                   float* __restrict__ Om,
                   int Qn, int Kn)
{
  // Double-buffered staging (statically distinct arrays -> no alias barriers)
  __shared__ __align__(16) _Float16 Kl0[CK * KST], Kl1[CK * KST];
  __shared__ __align__(16) _Float16 Vt0[DH * VST], Vt1[DH * VST];

  const int tid  = threadIdx.x;
  const int wave = tid >> 5;
  const int lane = tid & 31;
  const int n    = lane & 15;   // fragment column / query owned by this lane
  const int g    = lane >> 4;   // half-wave group

  const int b     = blockIdx.y;
  const int qbase = blockIdx.x * BQ;

  const int  vlen   = valid_lens[b];
  const int  qrow   = qbase + wave * 16 + n;
  const bool masked = (qrow >= vlen);

  const size_t baseQ = (size_t)b * Qn * DH;
  const size_t baseK = (size_t)b * Kn * DH;

  // ---- Q fragments (B-matrix of GEMM1). Pre-scaled by log2(e)/sqrt(D); the
  // masked-row trick: zero this lane's column -> S^T column is exactly 0.
  const float qscale = masked ? 0.0f
                     : 0.08838834764831845f * 1.44269504088896340736f;
  F16F qfrag[4];
  #pragma unroll
  for (int c = 0; c < 4; ++c) {
    const float* qp = Qm + baseQ + (size_t)qrow * DH + c * 32 + g * 8;
    float t[16];
    #pragma unroll
    for (int j = 0; j < 4; ++j) {
      float4 f = ((const float4*)qp)[(j & 1) + (j >> 1) * 4];
      t[4*j+0] = f.x; t[4*j+1] = f.y; t[4*j+2] = f.z; t[4*j+3] = f.w;
    }
    #pragma unroll
    for (int j = 0; j < 16; ++j) qfrag[c].v[j] = (_Float16)(t[j] * qscale);
  }

  // ---- running flash-softmax state (log2 domain) + O accumulators ----
  float m_run = -INFINITY;
  float l_run = 0.0f;
  v8f o[8];
  #pragma unroll
  for (int f = 0; f < 8; ++f) {
    #pragma unroll
    for (int r = 0; r < 8; ++r) o[f][r] = 0.0f;
  }

  // ---- staging: K chunk row-major f16, V chunk transposed f16.
  // Uniform base pointers + loop-invariant 32-bit lane offsets (GVS mode).
  auto stage = [&](int kb, _Float16* Kl, _Float16* Vt) {
    const float* kbp = Km + baseK + (size_t)kb * DH;   // uniform (SGPR base)
    const float* vbp = Vm + baseK + (size_t)kb * DH;   // uniform (SGPR base)
    #pragma unroll
    for (int it = 0; it < 4; ++it) {
      int idx = tid + it * 128;              // 0..511
      int key = idx >> 4;                    // 0..31
      int dc  = (idx & 15) * 8;              // 0..120
      int off = key * DH + dc;               // 32-bit lane offset
      float4 a0 = *(const float4*)(kbp + off);
      float4 a1 = *(const float4*)(kbp + off + 4);
      h8 hv;
      hv[0]=(_Float16)a0.x; hv[1]=(_Float16)a0.y; hv[2]=(_Float16)a0.z; hv[3]=(_Float16)a0.w;
      hv[4]=(_Float16)a1.x; hv[5]=(_Float16)a1.y; hv[6]=(_Float16)a1.z; hv[7]=(_Float16)a1.w;
      *(h8*)&Kl[key * KST + dc] = hv;
    }
    // d == tid (block has 128 threads); keys it*8..it*8+7 per sub-iteration.
    #pragma unroll
    for (int it = 0; it < 4; ++it) {
      h8 hv;
      #pragma unroll
      for (int j = 0; j < 8; ++j)            // coalesced across lanes (d contiguous)
        hv[j] = (_Float16)vbp[(it * 8 + j) * DH + tid];
      *(h8*)&Vt[tid * VST + it * 8] = hv;
    }
  };

  // Prefetch one 128B line per thread of K and V of a future chunk into L2.
  auto prefetchChunk = [&](int kb) {
    if (kb < Kn) {
      const float* kbp = Km + baseK + (size_t)kb * DH;
      const float* vbp = Vm + baseK + (size_t)kb * DH;
      __builtin_prefetch(kbp + tid * 32, 0, 1);
      __builtin_prefetch(vbp + tid * 32, 0, 1);
    }
  };

  auto computeChunk = [&](const _Float16* Kl, const _Float16* Vt) {
    // GEMM1: S^T tiles. A = K rows (16x32), B = Q^T, C = 16x16 f32.
    v8f st0, st1;
    #pragma unroll
    for (int r = 0; r < 8; ++r) { st0[r] = 0.0f; st1[r] = 0.0f; }
    #pragma unroll
    for (int c = 0; c < 4; ++c) {
      const int d0 = c * 32;
      // A layout: row = key = L%16; h<8 -> k = 8g+h; h>=8 -> k = 16+8g+(h-8)
      F16F a0, a1;
      a0.h[0] = *(const h8*)&Kl[(n)      * KST + d0 + 8 * g];
      a0.h[1] = *(const h8*)&Kl[(n)      * KST + d0 + 16 + 8 * g];
      a1.h[0] = *(const h8*)&Kl[(16 + n) * KST + d0 + 8 * g];
      a1.h[1] = *(const h8*)&Kl[(16 + n) * KST + d0 + 16 + 8 * g];
      st0 = __builtin_amdgcn_wmma_f32_16x16x32_f16(false, a0.v, false, qfrag[c].v,
                                                   (short)0, st0, false, false);
      st1 = __builtin_amdgcn_wmma_f32_16x16x32_f16(false, a1.v, false, qfrag[c].v,
                                                   (short)0, st1, false, false);
    }

    // Online softmax update in log2 domain (per-lane query = L%16).
    float mc = st0[0];
    #pragma unroll
    for (int r = 1; r < 8; ++r) mc = fmaxf(mc, st0[r]);
    #pragma unroll
    for (int r = 0; r < 8; ++r) mc = fmaxf(mc, st1[r]);
    mc = fmaxf(mc, __shfl_xor(mc, 16, 32));       // combine the two key groups
    const float m_new = fmaxf(m_run, mc);
    const float scl   = __builtin_amdgcn_exp2f(m_run - m_new); // iter0: exp2(-inf)=0
    m_run = m_new;

    float sc = 0.0f;
    F16F pa;                                      // S^T regs map 1:1 onto P A-frag
    #pragma unroll
    for (int r = 0; r < 8; ++r) {
      float p0 = __builtin_amdgcn_exp2f(st0[r] - m_new);  // keys kb + r + 8g
      float p1 = __builtin_amdgcn_exp2f(st1[r] - m_new);  // keys kb + 16 + r + 8g
      sc += p0 + p1;
      pa.v[r]     = (_Float16)p0;
      pa.v[8 + r] = (_Float16)p1;
    }
    sc += __shfl_xor(sc, 16, 32);
    l_run = l_run * scl + sc;

    // Broadcast rescale factor into O-fragment row layout (row = r + 8g).
    float s8[8];
    #pragma unroll
    for (int r = 0; r < 8; ++r) s8[r] = __shfl(scl, r + 8 * g, 32);
    #pragma unroll
    for (int f = 0; f < 8; ++f) {
      #pragma unroll
      for (int r = 0; r < 8; ++r) o[f][r] *= s8[r];
    }

    // GEMM2: O += P * V. B layout: col n = d, h<8 -> k = 8g+h; h>=8 -> k = 16+8g+(h-8).
    #pragma unroll
    for (int f = 0; f < 8; ++f) {
      const int d = f * 16 + n;
      F16F bv;
      bv.h[0] = *(const h8*)&Vt[d * VST + 8 * g];
      bv.h[1] = *(const h8*)&Vt[d * VST + 16 + 8 * g];
      o[f] = __builtin_amdgcn_wmma_f32_16x16x32_f16(false, pa.v, false, bv.v,
                                                    (short)0, o[f], false, false);
    }
  };

  // ---- software-pipelined main loop, unrolled x2 for static double buffering
  stage(0, Kl0, Vt0);
  prefetchChunk(CK);
  for (int kb = 0; kb < Kn; kb += 2 * CK) {
    __syncthreads();                      // buf0 stores of this round visible
    stage(kb + CK, Kl1, Vt1);             // fill buf1 while computing on buf0
    prefetchChunk(kb + 2 * CK);
    computeChunk(Kl0, Vt0);

    __syncthreads();                      // buf1 ready; buf0 reads done
    if (kb + 2 * CK < Kn) stage(kb + 2 * CK, Kl0, Vt0);
    prefetchChunk(kb + 3 * CK);
    computeChunk(Kl1, Vt1);
  }

  // ---- epilogue: O / l ; O-fragment row = r + 8g needs that query's 1/l ----
  const float li = 1.0f / l_run;
  float l8[8];
  #pragma unroll
  for (int r = 0; r < 8; ++r) l8[r] = __shfl(li, r + 8 * g, 32);

  #pragma unroll
  for (int f = 0; f < 8; ++f) {
    #pragma unroll
    for (int r = 0; r < 8; ++r) {
      const int qr = qbase + wave * 16 + r + 8 * g;
      Om[baseQ + (size_t)qr * DH + f * 16 + n] = o[f][r] * l8[r];
    }
  }
}

extern "C" void kernel_launch(void* const* d_in, const int* in_sizes, int n_in,
                              void* d_out, int out_size, void* d_ws, size_t ws_size,
                              hipStream_t stream) {
  const float* q  = (const float*)d_in[0];
  const float* k  = (const float*)d_in[1];
  const float* v  = (const float*)d_in[2];
  const int*   vl = (const int*)d_in[3];   // valid_lens (harness integer -> int32)
  float* out = (float*)d_out;

  const int B = 16, Qn = 2048, Kn = 2048;
  (void)in_sizes; (void)n_in; (void)out_size; (void)d_ws; (void)ws_size;

  dim3 grid(Qn / BQ, B);   // 32 query tiles x 16 batches
  dim3 block(128);         // 4 waves of 32
  attn_fwd_wmma<<<grid, block, 0, stream>>>(q, k, v, vl, out, Qn, Kn);
}
